// TransformerBlock2_7138235646707
// MI455X (gfx1250) — matrix operand order repaired and checked by reading, not compiled, not run on hardware
//
#include <hip/hip_runtime.h>
#include <hip/hip_bf16.h>
#include <math.h>

typedef __attribute__((ext_vector_type(16))) __bf16       v16bf;
typedef __attribute__((ext_vector_type(8)))  float        v8f;
typedef __attribute__((ext_vector_type(8)))  unsigned int v8u;
typedef __attribute__((ext_vector_type(4)))  unsigned int v4u;
typedef __attribute__((ext_vector_type(8)))  int          v8i;
typedef __attribute__((ext_vector_type(4)))  int          v4i;

union FragBF { v8u u; v16bf v; };

__device__ __forceinline__ unsigned short f2bf(float f) {
  unsigned int u = __float_as_uint(f);
  u += 0x7FFFu + ((u >> 16) & 1u);          // round-to-nearest-even
  return (unsigned short)(u >> 16);
}

__device__ __forceinline__ v8f wmma_bf16(const FragBF& a, const FragBF& b, v8f c) {
  return __builtin_amdgcn_wmma_f32_16x16x32_bf16(false, a.v, false, b.v,
                                                 (short)0, c, false, false);
}

// ---------------------------------------------------------------------------
// TDM: DMA a 2D bf16 tile [rows x rowElems] (row stride = strideElems) from
// global memory into LDS at lds_off, with LDS padding (padInterval code per
// ISA: interval = 2<<code DWORDs; padAmount code: amount = code+1 DWORDs).
__device__ __forceinline__ void tdm_load_2d(unsigned lds_off, const void* gptr,
                                            unsigned rowElems, unsigned rows,
                                            unsigned long long strideElems,
                                            unsigned padIntervalCode,
                                            unsigned padAmountCode) {
  unsigned long long ga = (unsigned long long)gptr;
  v4u g0 = {1u,                                   // count=1 (valid descriptor)
            lds_off,                              // lds_addr (bytes)
            (unsigned)ga,                         // global_addr[31:0]
            (unsigned)((ga >> 32) & 0x01FFFFFFu) | 0x80000000u}; // addr[56:32]|type=2
  int w0 = (int)((1u << 16) |                     // data_size = 1 (2 bytes)
                 (1u << 20) |                     // pad_enable
                 (padIntervalCode << 22) | (padAmountCode << 25));
  int w1 = (int)((rowElems & 0xFFFFu) << 16);     // tensor_dim0[15:0]
  int w2 = (int)((rowElems >> 16) | ((rows & 0xFFFFu) << 16));  // td0[31:16]|td1[15:0]
  int w3 = (int)((rows >> 16) | ((rowElems & 0xFFFFu) << 16));  // td1[31:16]|tile_dim0
  int w4 = (int)(rows & 0xFFFFu);                 // tile_dim1 (tile_dim2 = 0)
  int w5 = (int)(strideElems & 0xFFFFFFFFull);    // tensor_dim0_stride[31:0]
  int w6 = (int)((strideElems >> 32) & 0xFFFFull);// stride[47:32]
  int w7 = 0;
  v8i g1 = {w0, w1, w2, w3, w4, w5, w6, w7};
  v4i z4 = {0, 0, 0, 0};
#if defined(__clang_major__) && (__clang_major__ >= 23)
  v8i z8 = {0, 0, 0, 0, 0, 0, 0, 0};
  __builtin_amdgcn_tensor_load_to_lds(g0, g1, z4, z4, z8, 0);
#else
  __builtin_amdgcn_tensor_load_to_lds(g0, g1, z4, z4, 0);
#endif
}

__device__ __forceinline__ unsigned lds_offset_of(const void* p) {
  return (unsigned)(unsigned long long)(size_t)p;   // aperture in high 32 bits
}

// ---------------------------------------------------------------------------
// elementwise f32 -> bf16
__global__ __launch_bounds__(256) void k_cvt_bf16(const float* __restrict__ in,
                                                  unsigned short* __restrict__ out, int n) {
  int i = blockIdx.x * 256 + threadIdx.x;
  if (i < n) out[i] = f2bf(in[i]);
}

// W [K,N] f32 -> Wt [N,K] bf16 (coalesced writes)
__global__ __launch_bounds__(256) void k_transpose_bf16(const float* __restrict__ W,
                                                        unsigned short* __restrict__ Wt,
                                                        int K, int N) {
  long o = (long)blockIdx.x * 256 + threadIdx.x;
  if (o < (long)K * N) {
    int nn = (int)(o / K), kk = (int)(o % K);
    Wt[o] = f2bf(W[(size_t)kk * N + nn]);
  }
}

// per-row LayerNorm, f32 in -> bf16 out (block = 256, one row per block)
__global__ __launch_bounds__(256) void k_layernorm_bf16(
    const float* __restrict__ x, const float* __restrict__ g, const float* __restrict__ b,
    unsigned short* __restrict__ out, int D) {
  int row = blockIdx.x;
  const float* xr = x + (size_t)row * D;
  float s = 0.f, s2 = 0.f;
  for (int c = threadIdx.x; c < D; c += 256) { float v = xr[c]; s += v; s2 += v * v; }
  for (int m = 16; m; m >>= 1) { s += __shfl_xor(s, m, 32); s2 += __shfl_xor(s2, m, 32); }
  __shared__ float ps[8], ps2[8];
  if ((threadIdx.x & 31) == 0) { ps[threadIdx.x >> 5] = s; ps2[threadIdx.x >> 5] = s2; }
  __syncthreads();
  float tot = 0.f, tot2 = 0.f;
#pragma unroll
  for (int i = 0; i < 8; ++i) { tot += ps[i]; tot2 += ps2[i]; }
  float mu = tot / D;
  float var = tot2 / D - mu * mu;
  float rs = rsqrtf(var + 1e-5f);
  for (int c = threadIdx.x; c < D; c += 256)
    out[(size_t)row * D + c] = f2bf((xr[c] - mu) * rs * g[c] + b[c]);
}

// ---------------------------------------------------------------------------
// C[M,N] = A[M,K](bf16) * Bt[N,K](bf16)^T + bias, with fused epilogues.
// EPI: 0 = f32 store, 1 = f32 store + residual(res), 2 = GELU -> bf16,
//      3 = bf16 store, 4 = bf16 store transposed for V ([b,h,c,token])
// Block = 256 thr (8 waves). Block tile 128(M) x 64(N); wave tile 16 x 64.
// B tile (64 rows x 32 k) is TDM-DMA'd into LDS once per block, double
// buffered; pad 1 DWORD per 16 -> 34-short row pitch (conflict-free).
template <int EPI>
__global__ __launch_bounds__(256) void k_gemm(
    const unsigned short* __restrict__ A, const unsigned short* __restrict__ Bt,
    const float* __restrict__ bias, const float* __restrict__ res,
    float* __restrict__ outF, unsigned short* __restrict__ outB,
    int N, int K, int rowsPerBatch, int vtStride) {
  __shared__ __align__(16) unsigned short Bl[2][64 * 34];
  const int lane = threadIdx.x & 31;
  const int wave = threadIdx.x >> 5;
  const int l16  = lane & 15;
  const int hi   = lane >> 4;
  const int rowbase = blockIdx.y * 128 + wave * 16;
  const int colbase = blockIdx.x * 64;

  v8f zf = {0.f, 0.f, 0.f, 0.f, 0.f, 0.f, 0.f, 0.f};
  v8f acc[4];
#pragma unroll
  for (int t = 0; t < 4; ++t) acc[t] = zf;

  const unsigned short* Arow  = A + (size_t)(rowbase + l16) * K;
  const unsigned short* Bbase = Bt + (size_t)colbase * K;

  if (wave == 0)
    tdm_load_2d(lds_offset_of(&Bl[0][0]), Bbase, 32, 64, (unsigned)K, 3, 0);

  for (int k0 = 0; k0 < K; k0 += 32) {
    const int cur = (k0 >> 5) & 1;
    if (k0 + 32 < K) {
      if (wave == 0)
        tdm_load_2d(lds_offset_of(&Bl[cur ^ 1][0]), Bbase + k0 + 32, 32, 64,
                    (unsigned)K, 3, 0);
      __builtin_amdgcn_s_wait_tensorcnt(1);
    } else {
      __builtin_amdgcn_s_wait_tensorcnt(0);
    }
    __syncthreads();                         // B tile ready for all waves

    FragBF a;
#pragma unroll
    for (int r = 0; r < 8; ++r) {
      int c = k0 + hi * 16 + r * 2;   // A 16x32 layout (contiguous k = 16h + s)
      a.u[r] = *(const unsigned int*)(Arow + c);
    }
#pragma unroll
    for (int t = 0; t < 4; ++t) {
      FragBF bf;
      const unsigned short* Brow = &Bl[cur][(t * 16 + l16) * 34 + hi * 16];
#pragma unroll
      for (int r = 0; r < 8; ++r) bf.u[r] = *(const unsigned int*)(Brow + 2 * r);
      acc[t] = wmma_bf16(a, bf, acc[t]);
    }
    __syncthreads();                         // done reading before overwrite
  }

#pragma unroll
  for (int t = 0; t < 4; ++t) {
#pragma unroll
    for (int r = 0; r < 8; ++r) {
      int row = rowbase + r + hi * 8;          // C layout: comp r, half -> row
      int col = colbase + t * 16 + l16;
      float v = acc[t][r] + bias[col];
      if (EPI == 1) v += res[(size_t)row * N + col];
      if (EPI == 2) v = 0.5f * v * (1.0f + erff(v * 0.70710678118654752f));
      if (EPI == 0 || EPI == 1) {
        outF[(size_t)row * N + col] = v;
      } else if (EPI == 4) {
        int bb = row / rowsPerBatch, mm = row % rowsPerBatch;
        int h = col >> 6, c = col & 63;
        outB[((size_t)(bb * 8 + h) * 64 + c) * vtStride + mm] = f2bf(v);
      } else {
        outB[(size_t)row * N + col] = f2bf(v);
      }
    }
  }
}

// ---------------------------------------------------------------------------
// Flash attention: one wave per 16-query tile, 8 waves/block sharing (b,h).
// K/V 64-key tiles are TDM-DMA'd into LDS (double buffered) and shared by all
// waves; pad 2 DWORDs per 32 -> 68-short row pitch (conflict-free, 4B-aligned).
// Row-sums (l) via WMMA against an all-ones B fragment.
__global__ __launch_bounds__(256) void k_flash(
    const unsigned short* __restrict__ Q, const unsigned short* __restrict__ Kb,
    const unsigned short* __restrict__ Vt, unsigned short* __restrict__ O,
    int Nq, int Nkv, float scale) {
  __shared__ __align__(16) unsigned short KtL[2][64 * 68];
  __shared__ __align__(16) unsigned short VtL[2][64 * 68];
  __shared__ __align__(16) unsigned short Pl[8][16 * 64];
  const int lane = threadIdx.x & 31, wave = threadIdx.x >> 5;
  const int l16 = lane & 15, hi = lane >> 4;
  const int D = 512;
  const int qtPerBH = Nq >> 4;
  int w  = blockIdx.x * 8 + wave;
  int qt = w % qtPerBH;
  int h  = (w / qtPerBH) & 7;      // uniform across the block (qtPerBH % 8 == 0)
  int b  = w / (qtPerBH * 8);

  // Q fragments for both 32-wide contraction steps (d = 64)
  const unsigned short* Qrow = Q + (size_t)(b * Nq + qt * 16 + l16) * D + h * 64;
  FragBF qa[2];
#pragma unroll
  for (int s = 0; s < 2; ++s)
#pragma unroll
    for (int r = 0; r < 8; ++r) {
      int c = s * 32 + hi * 16 + r * 2;
      qa[s].u[r] = *(const unsigned int*)(Qrow + c);
    }

  FragBF ones;
#pragma unroll
  for (int r = 0; r < 8; ++r) ones.u[r] = 0x3F803F80u;   // bf16 1.0 pair

  v8f zf = {0.f, 0.f, 0.f, 0.f, 0.f, 0.f, 0.f, 0.f};
  v8f acc[4], lacc = zf;
#pragma unroll
  for (int t = 0; t < 4; ++t) acc[t] = zf;
  float m[8];
#pragma unroll
  for (int r = 0; r < 8; ++r) m[r] = -3.0e38f;

  unsigned short* P = &Pl[wave][0];
  const unsigned short* Kbase = Kb + (size_t)(b * Nkv) * D + h * 64;
  const unsigned short* Vbase = Vt + (size_t)((b * 8 + h) * 64) * Nkv;

  if (wave == 0) {
    tdm_load_2d(lds_offset_of(&KtL[0][0]), Kbase, 64, 64, (unsigned)D, 4, 1);
    tdm_load_2d(lds_offset_of(&VtL[0][0]), Vbase, 64, 64, (unsigned)Nkv, 4, 1);
  }

  for (int j = 0; j < Nkv; j += 64) {
    const int cur = (j >> 6) & 1;
    if (j + 64 < Nkv) {
      if (wave == 0) {
        tdm_load_2d(lds_offset_of(&KtL[cur ^ 1][0]), Kbase + (size_t)(j + 64) * D,
                    64, 64, (unsigned)D, 4, 1);
        tdm_load_2d(lds_offset_of(&VtL[cur ^ 1][0]), Vbase + (j + 64),
                    64, 64, (unsigned)Nkv, 4, 1);
      }
      __builtin_amdgcn_s_wait_tensorcnt(2);
    } else {
      __builtin_amdgcn_s_wait_tensorcnt(0);
    }
    __syncthreads();                       // K/V tiles ready

    const unsigned short* Kl = &KtL[cur][0];
    const unsigned short* Vl = &VtL[cur][0];

    // S[t]: scores for keys j+16t .. j+16t+15   (QK^T, contraction 64 = 2x32)
    v8f S[4];
#pragma unroll
    for (int t = 0; t < 4; ++t) S[t] = zf;
#pragma unroll
    for (int s = 0; s < 2; ++s) {
#pragma unroll
      for (int t = 0; t < 4; ++t) {
        FragBF kf;
        const unsigned short* Kr = Kl + (t * 16 + l16) * 68 + s * 32 + hi * 16;
#pragma unroll
        for (int r = 0; r < 8; ++r) kf.u[r] = *(const unsigned int*)(Kr + 2 * r);
        S[t] = wmma_bf16(qa[s], kf, S[t]);
      }
    }

    // online softmax per C-component (row = r + hi*8, keys across 16-lane half)
#pragma unroll
    for (int r = 0; r < 8; ++r) {
      float s0 = S[0][r] * scale, s1 = S[1][r] * scale;
      float s2 = S[2][r] * scale, s3 = S[3][r] * scale;
      float tm = fmaxf(fmaxf(s0, s1), fmaxf(s2, s3));
      for (int msk = 8; msk; msk >>= 1) tm = fmaxf(tm, __shfl_xor(tm, msk, 32));
      float mn = fmaxf(m[r], tm);
      float sc = expf(m[r] - mn);
      m[r] = mn;
      lacc[r] = lacc[r] * sc;
#pragma unroll
      for (int t = 0; t < 4; ++t) acc[t][r] = acc[t][r] * sc;
      int row = r + hi * 8;
      P[row * 64 + l16]      = f2bf(expf(s0 - mn));
      P[row * 64 + 16 + l16] = f2bf(expf(s1 - mn));
      P[row * 64 + 32 + l16] = f2bf(expf(s2 - mn));
      P[row * 64 + 48 + l16] = f2bf(expf(s3 - mn));
    }
    asm volatile("s_wait_dscnt 0" ::: "memory");

    // P as two A fragments (cols 0..31 and 32..63 of the 16x64 tile)
    FragBF pf0, pf1;
#pragma unroll
    for (int r = 0; r < 8; ++r) {
      int c = hi * 16 + r * 2;
      pf0.u[r] = *(const unsigned int*)(P + l16 * 64 + c);
      pf1.u[r] = *(const unsigned int*)(P + l16 * 64 + 32 + c);
    }

    // l += rowsum(P) via WMMA against ones
    lacc = wmma_bf16(pf0, ones, lacc);
    lacc = wmma_bf16(pf1, ones, lacc);

    // ctx += P * V   (two 32-key contraction steps per d-tile)
#pragma unroll
    for (int t = 0; t < 4; ++t) {
      FragBF vf0, vf1;
      const unsigned short* Vr = Vl + (t * 16 + l16) * 68 + hi * 16;
#pragma unroll
      for (int r = 0; r < 8; ++r) {
        vf0.u[r] = *(const unsigned int*)(Vr + 2 * r);
        vf1.u[r] = *(const unsigned int*)(Vr + 32 + 2 * r);
      }
      acc[t] = wmma_bf16(pf0, vf0, acc[t]);
      acc[t] = wmma_bf16(pf1, vf1, acc[t]);
    }
    __syncthreads();                       // done with tiles before overwrite
  }

#pragma unroll
  for (int r = 0; r < 8; ++r) {
    float inv = 1.0f / lacc[r];
    int row = qt * 16 + r + hi * 8;
#pragma unroll
    for (int t = 0; t < 4; ++t)
      O[(size_t)(b * Nq + row) * D + h * 64 + t * 16 + l16] = f2bf(acc[t][r] * inv);
  }
}

// ---------------------------------------------------------------------------
extern "C" void kernel_launch(void* const* d_in, const int* in_sizes, int n_in,
                              void* d_out, int out_size, void* d_ws, size_t ws_size,
                              hipStream_t stream) {
  const int B = 2, Nq = 2048, Nk = 4096, D = 512, Hid = 2048;
  const float* x     = (const float*)d_in[0];
  const float* cross = (const float*)d_in[1];
  const float* ca_wq = (const float*)d_in[2];  const float* ca_bq = (const float*)d_in[3];
  const float* ca_wk = (const float*)d_in[4];  const float* ca_bk = (const float*)d_in[5];
  const float* ca_wv = (const float*)d_in[6];  const float* ca_bv = (const float*)d_in[7];
  const float* ca_wo = (const float*)d_in[8];  const float* ca_bo = (const float*)d_in[9];
  const float* sa_wq = (const float*)d_in[10]; const float* sa_bq = (const float*)d_in[11];
  const float* sa_wk = (const float*)d_in[12]; const float* sa_bk = (const float*)d_in[13];
  const float* sa_wv = (const float*)d_in[14]; const float* sa_bv = (const float*)d_in[15];
  const float* sa_wo = (const float*)d_in[16]; const float* sa_bo = (const float*)d_in[17];
  const float* ln1_g = (const float*)d_in[18]; const float* ln1_b = (const float*)d_in[19];
  const float* ln2_g = (const float*)d_in[20]; const float* ln2_b = (const float*)d_in[21];
  const float* mlp_w1 = (const float*)d_in[22]; const float* mlp_b1 = (const float*)d_in[23];
  const float* mlp_w2 = (const float*)d_in[24]; const float* mlp_b2 = (const float*)d_in[25];

  char* ws = (char*)d_ws;
  size_t off = 0;
  auto carve = [&](size_t bytes) -> void* {
    void* p = ws + off; off += (bytes + 255) & ~(size_t)255; return p;
  };

  unsigned short* wq_ca = (unsigned short*)carve((size_t)D * D * 2);
  unsigned short* wk_ca = (unsigned short*)carve((size_t)D * D * 2);
  unsigned short* wv_ca = (unsigned short*)carve((size_t)D * D * 2);
  unsigned short* wo_ca = (unsigned short*)carve((size_t)D * D * 2);
  unsigned short* wq_sa = (unsigned short*)carve((size_t)D * D * 2);
  unsigned short* wk_sa = (unsigned short*)carve((size_t)D * D * 2);
  unsigned short* wv_sa = (unsigned short*)carve((size_t)D * D * 2);
  unsigned short* wo_sa = (unsigned short*)carve((size_t)D * D * 2);
  unsigned short* w_m1  = (unsigned short*)carve((size_t)Hid * D * 2);  // [2048,512]
  unsigned short* w_m2  = (unsigned short*)carve((size_t)D * Hid * 2);  // [512,2048]
  unsigned short* xb     = (unsigned short*)carve((size_t)B * Nq * D * 2);
  unsigned short* crossb = (unsigned short*)carve((size_t)B * Nk * D * 2); // + kb reused as h1
  unsigned short* kb     = (unsigned short*)carve((size_t)B * Nk * D * 2);
  unsigned short* qb     = (unsigned short*)carve((size_t)B * Nq * D * 2);
  unsigned short* vtb    = (unsigned short*)carve((size_t)B * Nk * D * 2);
  unsigned short* ctxb   = (unsigned short*)carve((size_t)B * Nq * D * 2);
  float*          xc     = (float*)carve((size_t)B * Nq * D * 4);
  unsigned short* xn     = (unsigned short*)carve((size_t)B * Nq * D * 2);
  // reuse
  unsigned short* q2   = qb;
  unsigned short* vt2  = vtb;
  unsigned short* k2   = vtb + (size_t)B * Nq * D;   // second half of vtb
  unsigned short* ctx2 = ctxb;
  unsigned short* yn   = xn;
  unsigned short* h1   = crossb;                     // 16 MB spanning crossb+kb
  float* y = (float*)d_out;

  auto tr = [&](const float* W, int K, int N, unsigned short* Wt) {
    int n = K * N;
    k_transpose_bf16<<<(n + 255) / 256, 256, 0, stream>>>(W, Wt, K, N);
  };
  tr(ca_wq, D, D, wq_ca); tr(ca_wk, D, D, wk_ca); tr(ca_wv, D, D, wv_ca); tr(ca_wo, D, D, wo_ca);
  tr(sa_wq, D, D, wq_sa); tr(sa_wk, D, D, wk_sa); tr(sa_wv, D, D, wv_sa); tr(sa_wo, D, D, wo_sa);
  tr(mlp_w1, D, Hid, w_m1); tr(mlp_w2, Hid, D, w_m2);

  { int n = B * Nq * D; k_cvt_bf16<<<(n + 255) / 256, 256, 0, stream>>>(x, xb, n); }
  { int n = B * Nk * D; k_cvt_bf16<<<(n + 255) / 256, 256, 0, stream>>>(cross, crossb, n); }

  auto gemm = [&](int epi, const unsigned short* A, const unsigned short* Bt,
                  const float* bias, const float* res, float* oF, unsigned short* oB,
                  int M, int N, int K, int rpb, int vstride) {
    dim3 grid(N / 64, M / 128), blk(256);
    switch (epi) {
      case 0: k_gemm<0><<<grid, blk, 0, stream>>>(A, Bt, bias, res, oF, oB, N, K, rpb, vstride); break;
      case 1: k_gemm<1><<<grid, blk, 0, stream>>>(A, Bt, bias, res, oF, oB, N, K, rpb, vstride); break;
      case 2: k_gemm<2><<<grid, blk, 0, stream>>>(A, Bt, bias, res, oF, oB, N, K, rpb, vstride); break;
      case 3: k_gemm<3><<<grid, blk, 0, stream>>>(A, Bt, bias, res, oF, oB, N, K, rpb, vstride); break;
      case 4: k_gemm<4><<<grid, blk, 0, stream>>>(A, Bt, bias, res, oF, oB, N, K, rpb, vstride); break;
    }
  };

  // --- cross attention ---
  gemm(3, xb,     wq_ca, ca_bq, nullptr, nullptr, qb,  B * Nq, D, D, 0, 0);
  gemm(3, crossb, wk_ca, ca_bk, nullptr, nullptr, kb,  B * Nk, D, D, 0, 0);
  gemm(4, crossb, wv_ca, ca_bv, nullptr, nullptr, vtb, B * Nk, D, D, Nk, Nk);
  k_flash<<<B * 8 * (Nq / 16) / 8, 256, 0, stream>>>(qb, kb, vtb, ctxb, Nq, Nk, 0.125f);
  gemm(0, ctxb, wo_ca, ca_bo, nullptr, xc, nullptr, B * Nq, D, D, 0, 0);

  // --- LN1 + self attention ---
  k_layernorm_bf16<<<B * Nq, 256, 0, stream>>>(xc, ln1_g, ln1_b, xn, D);
  gemm(3, xn, wq_sa, sa_bq, nullptr, nullptr, q2,  B * Nq, D, D, 0, 0);
  gemm(3, xn, wk_sa, sa_bk, nullptr, nullptr, k2,  B * Nq, D, D, 0, 0);
  gemm(4, xn, wv_sa, sa_bv, nullptr, nullptr, vt2, B * Nq, D, D, Nq, Nq);
  k_flash<<<B * 8 * (Nq / 16) / 8, 256, 0, stream>>>(q2, k2, vt2, ctx2, Nq, Nq, 0.125f);
  gemm(1, ctx2, wo_sa, sa_bo, xc, y, nullptr, B * Nq, D, D, 0, 0);   // y = xc + xs

  // --- LN2 + MLP + residual ---
  k_layernorm_bf16<<<B * Nq, 256, 0, stream>>>(y, ln2_g, ln2_b, yn, D);
  gemm(2, yn, w_m1, mlp_b1, nullptr, nullptr, h1, B * Nq, Hid, D, 0, 0);
  gemm(1, h1, w_m2, mlp_b2, y, y, nullptr, B * Nq, D, Hid, 0, 0);    // out = y + z
}
